// SaProtBackbone_31782757990835
// MI455X (gfx1250) — compile-verified, matrix operand order checked by reading, not run on hardware
//
#include <hip/hip_runtime.h>
#include <stdint.h>

// ---------------------------------------------------------------------------
// SaProt transformer backbone for MI455X (gfx1250, wave32, WMMA + TDM).
// Compute-bound (~1.1 TFLOP vs ~40us of HBM weight traffic at 23.3 TB/s):
//  - every matmul on v_wmma_f32_16x16x32_bf16 (fp32 accumulate)
//  - GEMM tiles delivered by the Tensor Data Mover (tensor_load_to_lds),
//    double-buffered in LDS, synchronized with s_wait_tensorcnt
//  - TDM descriptor pad fields give conflict-free LDS row pitch for free
//  - flash attention keeps the N x N score matrix out of HBM entirely
// ---------------------------------------------------------------------------

#define DEV __device__ __forceinline__

typedef __attribute__((ext_vector_type(16))) __bf16 v16bf;
typedef __attribute__((ext_vector_type(8)))  float  v8f;
typedef __attribute__((ext_vector_type(4)))  unsigned int v4u;
typedef __attribute__((ext_vector_type(8)))  int    v8i;
typedef __attribute__((ext_vector_type(4)))  int    v4i;

namespace cfg {
constexpr int B = 2, N = 1024, D = 1280, H = 20, L = 12, FF = 5120;
constexpr int HD = D / H;            // 64
constexpr int M  = B * N;            // 2048 rows
constexpr float SCALE = 0.125f;      // HD^-0.5
constexpr float EPS   = 1e-5f;
}

union BfPair { uint32_t u; __bf16 h[2]; };

DEV v8f wmma_bf16(v16bf a, v16bf b, v8f c) {
  // D = A(16x32 bf16) * B(32x16 bf16) + C(16x16 f32)
  return __builtin_amdgcn_wmma_f32_16x16x32_bf16(
      /*neg_a=*/false, a, /*neg_b=*/false, b,
      /*c_mod=*/(short)0, c, /*reuse_a=*/false, /*reuse_b=*/false);
}

DEV float gelu_exact(float x) {
  return 0.5f * x * (1.0f + erff(x * 0.7071067811865476f));
}

// Fragment loader: 16 "parallel" rows x 32 reduction values, reduction values
// contiguous (pairs loaded as packed b32). ISA 16-bit A 16x32 layout:
// lanes 0-15 hold K 0..7 / 16..23, lanes 16-31 hold K 8..15 / 24..31.
DEV v16bf frag_ld(const __bf16* base, int rstride, int lane) {
  const int r  = lane & 15;
  const int kb = (lane >> 4) << 3;           // 0 or 8
  const __bf16* p = base + (size_t)r * rstride + kb;
  v16bf f;
#pragma unroll
  for (int i = 0; i < 8; ++i) {
    const int kk = (i < 4) ? (2 * i) : (16 + 2 * (i - 4));
    BfPair bp; bp.u = *(const uint32_t*)(p + kk);
    f[2 * i]     = bp.h[0];
    f[2 * i + 1] = bp.h[1];
  }
  return f;
}

// B-fragment built from an fp32 [k][n] LDS tile (TDM-delivered W), converting
// to bf16 at fragment-build time. base points at (k=0, n-tile col 0).
DEV v16bf frag_ld_f32(const float* base, int rstride, int lane) {
  const int n  = lane & 15;
  const int kb = (lane >> 4) << 3;
  const float* p = base + n + (size_t)kb * rstride;
  v16bf f;
#pragma unroll
  for (int i = 0; i < 8; ++i) {
    const int kk = (i < 4) ? (2 * i) : (16 + 2 * (i - 4));
    f[2 * i]     = (__bf16)p[(size_t)kk * rstride];
    f[2 * i + 1] = (__bf16)p[(size_t)(kk + 1) * rstride];
  }
  return f;
}

// Fragment loader where the reduction dim is strided (used for V in P@V).
DEV v16bf frag_ld_strided(const __bf16* base, int kstride, int lane) {
  const int n  = lane & 15;
  const int kb = (lane >> 4) << 3;
  const __bf16* p = base + n + (size_t)kb * kstride;
  v16bf f;
#pragma unroll
  for (int i = 0; i < 8; ++i) {
    const int kk = (i < 4) ? (2 * i) : (16 + 2 * (i - 4));
    f[2 * i]     = p[(size_t)kk * kstride];
    f[2 * i + 1] = p[(size_t)(kk + 1) * kstride];
  }
  return f;
}

// ---------------------------------------------------------------------------
// TDM: issue a 2D tensor_load_to_lds. Descriptor per CDNA5 ISA:
//  group0: [1:0]=count=1, [63:32]=lds_addr, [120:64]=global_addr, [127:126]=2
//  group1: [17:16]=data_size, [20]=pad_enable, [24:22]=pad_interval,
//          [31:25]=pad_amount, [79:48]=tensor_dim0, [111:80]=tensor_dim1,
//          [127:112]=tile_dim0, [143:128]=tile_dim1, [207:160]=dim0_stride
//  groups 2/3: zero (2D tensor). Tracked by TENSORcnt.
// This toolchain's builtin takes 6 args (g0, g1, g2, g3, g4, cpol).
// ---------------------------------------------------------------------------
DEV void tdm_load_2d(uint32_t lds_addr, const void* gptr,
                     uint32_t data_size_code,      // 1 = 2B, 2 = 4B
                     uint32_t tile_d0, uint32_t tile_d1,
                     uint32_t row_stride_elems,
                     uint32_t pad_interval_code) { // pad 1 dword per row
  const uint64_t ga = (uint64_t)(uintptr_t)gptr;
  v4u g0;
  g0[0] = 1u;                                        // count = 1, user mode
  g0[1] = lds_addr;                                  // LDS byte address
  g0[2] = (uint32_t)ga;                              // global_addr[31:0]
  g0[3] = (uint32_t)((ga >> 32) & 0x01FFFFFFu) | (2u << 30);  // addr[56:32]|type=2
  v8i g1;
  g1[0] = (int)((data_size_code << 16) | (1u << 20) | (pad_interval_code << 22));
  g1[1] = (int)((tile_d0 & 0xFFFFu) << 16);          // tensor_dim0[15:0]
  g1[2] = (int)(((tile_d0 >> 16) & 0xFFFFu) | ((tile_d1 & 0xFFFFu) << 16));
  g1[3] = (int)(((tile_d1 >> 16) & 0xFFFFu) | ((tile_d0 & 0xFFFFu) << 16)); // tile_dim0
  g1[4] = (int)(tile_d1 & 0xFFFFu);                  // tile_dim1 (tile_dim2 = 0)
  g1[5] = (int)row_stride_elems;                     // tensor_dim0_stride[31:0]
  g1[6] = 0;
  g1[7] = 0;
  const v4i gz4 = {0, 0, 0, 0};
  const v8i gz8 = {0, 0, 0, 0, 0, 0, 0, 0};
  __builtin_amdgcn_tensor_load_to_lds(g0, g1, gz4, gz4, gz8, 0);
}

// ---------------------------------------------------------------------------
// Embedding gather: X[b,n,:] = aa_emb[aa] + fs_emb[fs] + pos_emb[n]
// ---------------------------------------------------------------------------
__global__ __launch_bounds__(256) void embed_kernel(
    const int* __restrict__ aa, const int* __restrict__ fs,
    const float* __restrict__ aaE, const float* __restrict__ fsE,
    const float* __restrict__ posE, float* __restrict__ X) {
  using namespace cfg;
  const int t = blockIdx.x;            // token 0..M-1
  const int n = t % N;
  const int a = aa[t], f = fs[t];
#pragma unroll
  for (int i = 0; i < 5; ++i) {
    const int d = threadIdx.x + i * 256;   // D = 1280 = 5*256
    X[(size_t)t * D + d] =
        aaE[(size_t)a * D + d] + fsE[(size_t)f * D + d] + posE[(size_t)n * D + d];
  }
}

// ---------------------------------------------------------------------------
// Row LayerNorm. OUTF32=0 -> bf16 activations, OUTF32=1 -> fp32 (final LN).
// ---------------------------------------------------------------------------
template <int OUTF32>
__global__ __launch_bounds__(256) void ln_kernel(
    const float* __restrict__ X, const float* __restrict__ g,
    const float* __restrict__ b, void* __restrict__ out) {
  using namespace cfg;
  __shared__ float red[256];
  const int row = blockIdx.x, tid = threadIdx.x;
  const float* x = X + (size_t)row * D;
  float vals[5], s = 0.f, sq = 0.f;
#pragma unroll
  for (int i = 0; i < 5; ++i) {
    const float v = x[tid + i * 256];
    vals[i] = v; s += v; sq += v * v;
  }
  red[tid] = s; __syncthreads();
  for (int off = 128; off > 0; off >>= 1) {
    if (tid < off) red[tid] += red[tid + off];
    __syncthreads();
  }
  const float mean = red[0] * (1.0f / D);
  __syncthreads();
  red[tid] = sq; __syncthreads();
  for (int off = 128; off > 0; off >>= 1) {
    if (tid < off) red[tid] += red[tid + off];
    __syncthreads();
  }
  const float var  = red[0] * (1.0f / D) - mean * mean;
  const float rstd = rsqrtf(var + EPS);
#pragma unroll
  for (int i = 0; i < 5; ++i) {
    const int d = tid + i * 256;
    const float o = (vals[i] - mean) * rstd * g[d] + b[d];
    if constexpr (OUTF32) ((float*)out)[(size_t)row * D + d] = o;
    else                  ((__bf16*)out)[(size_t)row * D + d] = (__bf16)o;
  }
}

// ---------------------------------------------------------------------------
// WMMA GEMM with TDM double buffering.
// C[M,Nc] = A[M,K](bf16) * W[K,Nc](fp32 -> bf16 at frag build) + bias
// Block 256 threads = 8 waves, 128x128 tile, BK=32.
// Wave 0 DMAs stage s+1 (2 tensor_load_to_lds), waits TENSORcnt<=2 so stage s
// is resident, barrier, all 8 waves run 64 WMMAs on stage s.
// TDM pad: A rows 64B + 4B pad -> pitch 34 bf16; W rows 512B + 4B -> 129 f32.
// ---------------------------------------------------------------------------
enum { EPI_BF16 = 0, EPI_BF16_GELU = 1, EPI_F32_RESID = 2 };

template <int EPI>
__global__ __launch_bounds__(256, 2) void wmma_gemm_kernel(
    const __bf16* __restrict__ A, const float* __restrict__ W,
    const float* __restrict__ bias, void* __restrict__ Cout,
    const float* __restrict__ resid, int K, int Nc) {
  __shared__ __bf16 As[2][128][34];   // TDM-padded pitch: 2 stages, 17408 B
  __shared__ float  Wsf[2][32][129];  // TDM-padded pitch: 2 stages, 33024 B

  const int tid  = threadIdx.x;
  const int lane = tid & 31, wid = tid >> 5;
  const int wm = (wid >> 2) * 64;          // wave row offset in tile
  const int wn = (wid & 3) * 32;           // wave col offset in tile
  const int bm = blockIdx.y * 128, bn = blockIdx.x * 128;

  const uint32_t asAddr = (uint32_t)(uintptr_t)&As[0][0][0];
  const uint32_t wsAddr = (uint32_t)(uintptr_t)&Wsf[0][0][0];
  constexpr uint32_t AS_STAGE = 128u * 34u * 2u;   // 8704 B
  constexpr uint32_t WS_STAGE = 32u * 129u * 4u;   // 16512 B

  const v8f vzero = {0.f, 0.f, 0.f, 0.f, 0.f, 0.f, 0.f, 0.f};
  v8f acc[4][2];
#pragma unroll
  for (int i = 0; i < 4; ++i)
#pragma unroll
    for (int j = 0; j < 2; ++j) acc[i][j] = vzero;

  auto issue_stage = [&](int step, int st) {
    const int k0 = step * 32;
    // A tile: 128 rows x 32 bf16, row pitch K; pad every 16 dwords (code 3)
    tdm_load_2d(asAddr + (uint32_t)st * AS_STAGE,
                A + (size_t)bm * K + k0, 1u, 32u, 128u, (uint32_t)K, 3u);
    // W tile: 32 rows x 128 f32, row pitch Nc; pad every 128 dwords (code 6)
    tdm_load_2d(wsAddr + (uint32_t)st * WS_STAGE,
                W + (size_t)k0 * Nc + bn, 2u, 128u, 32u, (uint32_t)Nc, 6u);
  };

  const int nsteps = K / 32;
  if (wid == 0) issue_stage(0, 0);

  for (int s = 0; s < nsteps; ++s) {
    const int st = s & 1;
    if (wid == 0) {
      if (s + 1 < nsteps) {
        issue_stage(s + 1, st ^ 1);
        __builtin_amdgcn_s_wait_tensorcnt(2);   // stage st resident
      } else {
        __builtin_amdgcn_s_wait_tensorcnt(0);
      }
    }
    __syncthreads();

    v16bf af[4], bfr[2];
#pragma unroll
    for (int mt = 0; mt < 4; ++mt)
      af[mt] = frag_ld(&As[st][wm + mt * 16][0], 34, lane);
#pragma unroll
    for (int nt = 0; nt < 2; ++nt)
      bfr[nt] = frag_ld_f32(&Wsf[st][0][wn + nt * 16], 129, lane);
#pragma unroll
    for (int mt = 0; mt < 4; ++mt)
#pragma unroll
      for (int nt = 0; nt < 2; ++nt)
        acc[mt][nt] = wmma_bf16(af[mt], bfr[nt], acc[mt][nt]);
    __syncthreads();   // stage st free for the DMA issued next iteration
  }

  // ---- epilogue (C layout: vgpr r -> row r | r+8, col = lane&15) ----
  const int half = lane >> 4, lc = lane & 15;
#pragma unroll
  for (int mt = 0; mt < 4; ++mt)
#pragma unroll
    for (int nt = 0; nt < 2; ++nt)
#pragma unroll
      for (int r = 0; r < 8; ++r) {
        const int row = bm + wm + mt * 16 + r + half * 8;
        const int col = bn + wn + nt * 16 + lc;
        const size_t idx = (size_t)row * Nc + col;
        const float v = acc[mt][nt][r] + bias[col];
        if constexpr (EPI == EPI_BF16)
          ((__bf16*)Cout)[idx] = (__bf16)v;
        else if constexpr (EPI == EPI_BF16_GELU)
          ((__bf16*)Cout)[idx] = (__bf16)gelu_exact(v);
        else
          ((float*)Cout)[idx] = resid[idx] + v;
      }
}

// ---------------------------------------------------------------------------
// Flash attention: block = 4 waves, each wave one 16-row q tile of (b,h).
// Per iteration: 32 keys -> 4 score WMMAs, online softmax, P via LDS, 4 P@V
// WMMAs. Scores never leave registers/LDS.
// ---------------------------------------------------------------------------
__global__ __launch_bounds__(128, 4) void attn_kernel(
    const __bf16* __restrict__ Qb, const __bf16* __restrict__ Kb,
    const __bf16* __restrict__ Vb, const unsigned char* __restrict__ mask,
    __bf16* __restrict__ Ob) {
  using namespace cfg;
  __shared__ __bf16 Pbuf[4][16][36];   // per-wave P tile (16 q x 32 keys)

  const int lane = threadIdx.x & 31, wid = threadIdx.x >> 5;
  const int half = lane >> 4, lc = lane & 15;
  const int q0 = blockIdx.x * 64 + wid * 16;
  const int hy = blockIdx.y;          // head
  const int bz = blockIdx.z;          // batch
  const size_t rowbase = (size_t)bz * N;

  // Q fragments: 16 rows x 64 head-dim = 2 K-fragments, loaded once.
  v16bf qf[2];
#pragma unroll
  for (int kf = 0; kf < 2; ++kf)
    qf[kf] = frag_ld(Qb + (rowbase + q0) * D + hy * HD + kf * 32, D, lane);

  const v8f vzero = {0.f, 0.f, 0.f, 0.f, 0.f, 0.f, 0.f, 0.f};
  float m_i[8], l_i[8];
  v8f o_acc[4];
#pragma unroll
  for (int r = 0; r < 8; ++r) { m_i[r] = -1e30f; l_i[r] = 0.f; }
#pragma unroll
  for (int nt = 0; nt < 4; ++nt) o_acc[nt] = vzero;

  for (int k0 = 0; k0 < N; k0 += 32) {
    // S = Q * K^T for 2 key tiles of 16
    v8f s0 = vzero, s1 = vzero;
#pragma unroll
    for (int kf = 0; kf < 2; ++kf) {
      const v16bf k0f = frag_ld(Kb + (rowbase + k0) * D + hy * HD + kf * 32, D, lane);
      const v16bf k1f = frag_ld(Kb + (rowbase + k0 + 16) * D + hy * HD + kf * 32, D, lane);
      s0 = wmma_bf16(qf[kf], k0f, s0);
      s1 = wmma_bf16(qf[kf], k1f, s1);
    }
    const unsigned char mb0 = mask[rowbase + k0 + lc];
    const unsigned char mb1 = mask[rowbase + k0 + 16 + lc];

#pragma unroll
    for (int r = 0; r < 8; ++r) {
      const float a0 = mb0 ? s0[r] * SCALE : -1e30f;
      const float a1 = mb1 ? s1[r] * SCALE : -1e30f;
      float rm = fmaxf(a0, a1);
#pragma unroll
      for (int mk = 1; mk < 16; mk <<= 1) rm = fmaxf(rm, __shfl_xor(rm, mk, 32));
      const float mn = fmaxf(m_i[r], rm);
      const float alpha = __expf(m_i[r] - mn);
      const float p0 = __expf(a0 - mn), p1 = __expf(a1 - mn);
      float rs = p0 + p1;
#pragma unroll
      for (int mk = 1; mk < 16; mk <<= 1) rs += __shfl_xor(rs, mk, 32);
      l_i[r] = l_i[r] * alpha + rs;
      m_i[r] = mn;
      const int rr = r + half * 8;
      Pbuf[wid][rr][lc]      = (__bf16)p0;
      Pbuf[wid][rr][16 + lc] = (__bf16)p1;
#pragma unroll
      for (int nt = 0; nt < 4; ++nt) o_acc[nt][r] *= alpha;
    }
    __syncthreads();

    // O += P * V (K = 32 keys, N = 64 head-dim as 4 tiles)
    const v16bf pf = frag_ld(&Pbuf[wid][0][0], 36, lane);
#pragma unroll
    for (int nt = 0; nt < 4; ++nt) {
      const v16bf vf = frag_ld_strided(
          Vb + (rowbase + k0) * D + hy * HD + nt * 16, D, lane);
      o_acc[nt] = wmma_bf16(pf, vf, o_acc[nt]);
    }
    __syncthreads();
  }

#pragma unroll
  for (int r = 0; r < 8; ++r) {
    const float inv = (l_i[r] > 0.f) ? 1.f / l_i[r] : 0.f;
    const int row = q0 + r + half * 8;
#pragma unroll
    for (int nt = 0; nt < 4; ++nt)
      Ob[(rowbase + row) * D + hy * HD + nt * 16 + lc] =
          (__bf16)(o_acc[nt][r] * inv);
  }
}

// ---------------------------------------------------------------------------
// Host driver: 12 layers x (LN, QKV, attn, O+res, LN, FFN1+GELU, FFN2+res)
// ---------------------------------------------------------------------------
extern "C" void kernel_launch(void* const* d_in, const int* in_sizes, int n_in,
                              void* d_out, int out_size, void* d_ws, size_t ws_size,
                              hipStream_t stream) {
  using namespace cfg;
  (void)in_sizes; (void)n_in; (void)out_size; (void)ws_size;

  const int* aa_tok = (const int*)d_in[0];
  const int* fs_tok = (const int*)d_in[1];
  const unsigned char* mask = (const unsigned char*)d_in[2];
  const float* aaE  = (const float*)d_in[3];
  const float* fsE  = (const float*)d_in[4];
  const float* posE = (const float*)d_in[5];
  const float* ln1g = (const float*)d_in[6];
  const float* ln1b = (const float*)d_in[7];
  const float* wq = (const float*)d_in[8];
  const float* bq = (const float*)d_in[9];
  const float* wk = (const float*)d_in[10];
  const float* bk = (const float*)d_in[11];
  const float* wv = (const float*)d_in[12];
  const float* bv = (const float*)d_in[13];
  const float* wo = (const float*)d_in[14];
  const float* bo = (const float*)d_in[15];
  const float* ln2g = (const float*)d_in[16];
  const float* ln2b = (const float*)d_in[17];
  const float* w1 = (const float*)d_in[18];
  const float* b1 = (const float*)d_in[19];
  const float* w2 = (const float*)d_in[20];
  const float* b2 = (const float*)d_in[21];
  const float* flng = (const float*)d_in[22];
  const float* flnb = (const float*)d_in[23];

  char* p = (char*)d_ws;
  auto carve = [&](size_t bytes) {
    char* r = p;
    p += (bytes + 255) & ~(size_t)255;
    return r;
  };
  float*  X  = (float*) carve((size_t)M * D * sizeof(float));
  __bf16* Hb = (__bf16*)carve((size_t)M * D * sizeof(__bf16));
  __bf16* Qb = (__bf16*)carve((size_t)M * D * sizeof(__bf16));
  __bf16* Kb = (__bf16*)carve((size_t)M * D * sizeof(__bf16));
  __bf16* Vb = (__bf16*)carve((size_t)M * D * sizeof(__bf16));
  __bf16* Ob = (__bf16*)carve((size_t)M * D * sizeof(__bf16));
  __bf16* Fb = (__bf16*)carve((size_t)M * FF * sizeof(__bf16));

  embed_kernel<<<M, 256, 0, stream>>>(aa_tok, fs_tok, aaE, fsE, posE, X);

  const dim3 gD(D / 128, M / 128);      // 10 x 16
  const dim3 gFF(FF / 128, M / 128);    // 40 x 16
  const dim3 gAttn(N / 64, H, B);       // 16 x 20 x 2

  for (int l = 0; l < L; ++l) {
    const size_t oDD = (size_t)l * D * D;
    const size_t oD  = (size_t)l * D;
    const size_t oDF = (size_t)l * D * FF;
    const size_t oF  = (size_t)l * FF;

    ln_kernel<0><<<M, 256, 0, stream>>>(X, ln1g + oD, ln1b + oD, Hb);
    wmma_gemm_kernel<EPI_BF16><<<gD, 256, 0, stream>>>(Hb, wq + oDD, bq + oD, Qb, nullptr, D, D);
    wmma_gemm_kernel<EPI_BF16><<<gD, 256, 0, stream>>>(Hb, wk + oDD, bk + oD, Kb, nullptr, D, D);
    wmma_gemm_kernel<EPI_BF16><<<gD, 256, 0, stream>>>(Hb, wv + oDD, bv + oD, Vb, nullptr, D, D);
    attn_kernel<<<gAttn, 128, 0, stream>>>(Qb, Kb, Vb, mask, Ob);
    wmma_gemm_kernel<EPI_F32_RESID><<<gD, 256, 0, stream>>>(Ob, wo + oDD, bo + oD, X, X, D, D);
    ln_kernel<0><<<M, 256, 0, stream>>>(X, ln2g + oD, ln2b + oD, Hb);
    wmma_gemm_kernel<EPI_BF16_GELU><<<gFF, 256, 0, stream>>>(Hb, w1 + oDF, b1 + oF, Fb, nullptr, D, FF);
    wmma_gemm_kernel<EPI_F32_RESID><<<gD, 256, 0, stream>>>(Fb, w2 + oDF, b2 + oD, X, X, FF, D);
  }
  ln_kernel<1><<<M, 256, 0, stream>>>(X, flng, flnb, d_out);
}
